// RelationalGNN_75247827026322
// MI455X (gfx1250) — matrix operand order repaired and checked by reading, not compiled, not run on hardware
//
#include <hip/hip_runtime.h>
#include <hip/hip_bf16.h>

// ---------------------------------------------------------------------------
// RelationalGNN (hetero SAGE, 3 layers) for MI455X / gfx1250 (wave32, WMMA)
// ---------------------------------------------------------------------------

#define N_USER 100000
#define N_PROD 20000
#define N_EDGE 600000
#define D_FEAT 128   // din for every aggregation step

typedef __bf16 bf16_t;
typedef __attribute__((ext_vector_type(16))) __bf16 v16bf;
typedef __attribute__((ext_vector_type(8)))  float  v8f;

// ---------------------------------------------------------------------------
// Scatter-add aggregation: one 128-thread group per edge (dim = 128 always).
// agg[dst,:] += x_src[src,:] ; cnt[dst] += 1      (float atomics -> L2)
// ---------------------------------------------------------------------------
__global__ void gnn_scatter_add(const float* __restrict__ xsrc,
                                const int*   __restrict__ edge,   // [2, E] flat
                                int E,
                                float* __restrict__ agg,
                                float* __restrict__ cnt)
{
    const int epb = blockDim.x / D_FEAT;                 // edges per block (2)
    const int le  = threadIdx.x / D_FEAT;
    const int j   = threadIdx.x % D_FEAT;
    const long long e = (long long)blockIdx.x * epb + le;
    if (e >= E) return;
    const int src = edge[e];
    const int dst = edge[(size_t)E + e];
    atomicAdd(&agg[(size_t)dst * D_FEAT + j], xsrc[(size_t)src * D_FEAT + j]);
    if (j == 0) atomicAdd(&cnt[dst], 1.0f);
}

// agg[i,:] /= max(cnt[i], 1)
__global__ void gnn_mean_div(float* __restrict__ agg,
                             const float* __restrict__ cnt,
                             long long total /* n*dim */)
{
    long long i = (long long)blockIdx.x * blockDim.x + threadIdx.x;
    if (i >= total) return;
    float c = cnt[i / D_FEAT];
    agg[i] = agg[i] / fmaxf(c, 1.0f);
}

// one-shot weight conversion f32 -> bf16 (weights are loop-invariant & tiny)
__global__ void gnn_f32_to_bf16(const float* __restrict__ in,
                                bf16_t* __restrict__ out, int n)
{
    int i = blockIdx.x * blockDim.x + threadIdx.x;
    if (i < n) out[i] = (bf16_t)in[i];
}

// ---------------------------------------------------------------------------
// Fused WMMA GEMM:  out = act( A @ W1^T  [+ A2 @ W2^T]  + bias )
//   A/A2 : [M,K] f32 row-major;  W1/W2 : [N,K] bf16 row-major (preconverted)
// Each wave computes a 16 x 64 strip (NT=4): A fragment loaded once per
// K-step, B fragments are direct 32B v16bf loads into distinct registers,
// K fully unrolled (template) so loads of step k+1 overlap WMMAs of step k.
// ---------------------------------------------------------------------------
#define NT 4

__device__ __forceinline__ v16bf load_a_frag(const float* __restrict__ A,
                                             size_t rowOff, int k0, int kaOff)
{
    const float* ap = A + rowOff + k0 + kaOff;
    v16bf a;
#pragma unroll
    for (int i = 0; i < 16; ++i) {
        a[i]     = (bf16_t)ap[i];        // K = k0+kaOff    .. +15
    }
    return a;
}

template <int KSZ, bool DUAL>
__global__ void gnn_gemm_wmma(const float*  __restrict__ A,
                              const bf16_t* __restrict__ W1,
                              const float*  __restrict__ A2,   // used iff DUAL
                              const bf16_t* __restrict__ W2,   // used iff DUAL
                              const float*  __restrict__ bias,
                              float* __restrict__ out,
                              int M, int N, int relu)
{
    const int lane   = threadIdx.x & 31;
    const int wib    = threadIdx.x >> 5;
    const int nStrip = N / (16 * NT);
    const int strips = (M >> 4) * nStrip;
    const int wid    = blockIdx.x * (blockDim.x >> 5) + wib;
    if (wid >= strips) return;                // wave-uniform: EXEC stays all-1s

    const int mBase = (wid / nStrip) << 4;
    const int nBase = (wid % nStrip) * (16 * NT);

    const int l15   = lane & 15;
    const int hi    = lane >> 4;              // 0 | 1
    const int kaOff = hi ? 16 : 0;            // A half-lane K shift
    const int kbOff = hi ? 16 : 0;            // B half-lane K shift

    const size_t aRow = (size_t)(mBase + l15) * KSZ;    // A row for this lane
    size_t wRow[NT];
#pragma unroll
    for (int t = 0; t < NT; ++t)
        wRow[t] = (size_t)(nBase + 16 * t + l15) * KSZ; // W row == B column

    v8f c[NT];
#pragma unroll
    for (int t = 0; t < NT; ++t) c[t] = (v8f){};

#pragma unroll
    for (int k0 = 0; k0 < KSZ; k0 += 32) {
        // ---- neighbor-term: A @ W1^T ----
        const v16bf a1 = load_a_frag(A, aRow, k0, kaOff);
        v16bf b1[NT];
#pragma unroll
        for (int t = 0; t < NT; ++t)
            b1[t] = *(const v16bf*)(W1 + wRow[t] + k0 + kbOff);  // 32B aligned
#pragma unroll
        for (int t = 0; t < NT; ++t)
            c[t] = __builtin_amdgcn_wmma_f32_16x16x32_bf16(
                       false, a1, false, b1[t], (short)0, c[t], false, false);

        if (DUAL) {
            // ---- root-term: A2 @ W2^T (fused in same K-step) ----
            const v16bf a2 = load_a_frag(A2, aRow, k0, kaOff);
            v16bf b2[NT];
#pragma unroll
            for (int t = 0; t < NT; ++t)
                b2[t] = *(const v16bf*)(W2 + wRow[t] + k0 + kbOff);
#pragma unroll
            for (int t = 0; t < NT; ++t)
                c[t] = __builtin_amdgcn_wmma_f32_16x16x32_bf16(
                           false, a2, false, b2[t], (short)0, c[t], false, false);
        }
    }

#pragma unroll
    for (int t = 0; t < NT; ++t) {
        const int col  = nBase + 16 * t + l15;
        const float bv = bias[col];
#pragma unroll
        for (int r = 0; r < 8; ++r) {
            const int row = mBase + r + (hi ? 8 : 0);  // C layout: VGPR r -> M=r / r+8
            float v = c[t][r] + bv;
            if (relu) v = v > 0.0f ? v : 0.0f;
            out[(size_t)row * N + col] = v;
        }
    }
}

// ---------------------------------------------------------------------------
// Host-side orchestration (graph-capture safe: only async ops on `stream`)
// ---------------------------------------------------------------------------
static inline size_t alignUp(size_t x, size_t a) { return (x + a - 1) & ~(a - 1); }

static void launch_gemm(const float* A, const bf16_t* W1,
                        const float* A2, const bf16_t* W2,
                        const float* bias, float* out,
                        int M, int K, int N, int relu, hipStream_t stream)
{
    const int strips = (M >> 4) * (N >> 6);       // NT=4 -> 16x64 strips
    const int blocks = (strips + 7) / 8;          // 8 waves (256 thr) per block
    if (K == 128 && A2 != nullptr)
        gnn_gemm_wmma<128, true ><<<blocks, 256, 0, stream>>>(A, W1, A2, W2, bias, out, M, N, relu);
    else if (K == 128)
        gnn_gemm_wmma<128, false><<<blocks, 256, 0, stream>>>(A, W1, nullptr, nullptr, bias, out, M, N, relu);
    else
        gnn_gemm_wmma<64,  false><<<blocks, 256, 0, stream>>>(A, W1, nullptr, nullptr, bias, out, M, N, relu);
}

extern "C" void kernel_launch(void* const* d_in, const int* in_sizes, int n_in,
                              void* d_out, int out_size, void* d_ws, size_t ws_size,
                              hipStream_t stream)
{
    const float* x_user = (const float*)d_in[0];
    const float* x_prod = (const float*)d_in[1];
    const int*   e_u2p  = (const int*)d_in[2];    // [2, E]: src=user,  dst=prod
    const int*   e_p2u  = (const int*)d_in[3];    // [2, E]: src=prod,  dst=user

    // params: 10 arrays per layer, starting at d_in[4]
    //  0:u2p.l.W 1:u2p.l.b 2:u2p.r.W 3:p2u.l.W 4:p2u.l.b 5:p2u.r.W
    //  6:T_u2p.W 7:T_u2p.b 8:T_p2u.W 9:T_p2u.b
    auto P = [&](int layer, int idx) -> const float* {
        return (const float*)d_in[4 + layer * 10 + idx];
    };

    // ---- workspace layout (256B aligned) ----
    char* ws = (char*)d_ws;
    size_t off = 0;
    auto carveF = [&](size_t floats) -> float* {
        float* p = (float*)(ws + off);
        off = alignUp(off + floats * sizeof(float), 256);
        return p;
    };
    auto carveB = [&](size_t elems) -> bf16_t* {
        bf16_t* p = (bf16_t*)(ws + off);
        off = alignUp(off + elems * sizeof(bf16_t), 256);
        return p;
    };
    float* aggU = carveF((size_t)N_USER * D_FEAT);
    float* aggP = carveF((size_t)N_PROD * D_FEAT);
    float* cntU = carveF(N_USER);
    float* cntP = carveF(N_PROD);
    const size_t zeroBytes = off;            // contiguous agg+cnt region
    float* U0 = carveF((size_t)N_USER * D_FEAT);
    float* U2 = carveF((size_t)N_USER * D_FEAT);
    float* P0 = carveF((size_t)N_PROD * D_FEAT);
    float* P2 = carveF((size_t)N_PROD * D_FEAT);

    // bf16 copies of the 6 weight matrices of each layer (indices 0,2,3,5,6,8)
    bf16_t* Wb[3][10] = {};
    for (int l = 0; l < 3; ++l) {
        const int dout = (l == 2) ? 64 : 128;
        const int wIdx[6]  = {0, 2, 3, 5, 6, 8};
        const int wSize[6] = {dout * D_FEAT, dout * D_FEAT, dout * D_FEAT,
                              dout * D_FEAT, dout * dout,   dout * dout};
        for (int q = 0; q < 6; ++q) {
            Wb[l][wIdx[q]] = carveB(wSize[q]);
            gnn_f32_to_bf16<<<(wSize[q] + 255) / 256, 256, 0, stream>>>(
                P(l, wIdx[q]), Wb[l][wIdx[q]], wSize[q]);
        }
    }

    float* outU_final = (float*)d_out;                          // [N_USER,64]
    float* outP_final = (float*)d_out + (size_t)N_USER * 64;    // [N_PROD,64]

    const float* xu_in = x_user;
    const float* xp_in = x_prod;

    const int scatterBlocks = (N_EDGE + 1) / 2;   // 2 edges per 256-thread block

    for (int layer = 0; layer < 3; ++layer) {
        const int dout = (layer == 2) ? 64 : 128;
        const int relu = (layer < 2) ? 1 : 0;

        // 1) zero aggregation buffers + counts
        hipMemsetAsync(d_ws, 0, zeroBytes, stream);

        // 2) scatter-add both relations (din is always 128)
        gnn_scatter_add<<<scatterBlocks, 256, 0, stream>>>(
            xu_in, e_u2p, N_EDGE, aggP, cntP);
        gnn_scatter_add<<<scatterBlocks, 256, 0, stream>>>(
            xp_in, e_p2u, N_EDGE, aggU, cntU);

        // 3) mean = agg / max(cnt,1)
        {
            long long tP = (long long)N_PROD * D_FEAT;
            long long tU = (long long)N_USER * D_FEAT;
            gnn_mean_div<<<(int)((tP + 255) / 256), 256, 0, stream>>>(aggP, cntP, tP);
            gnn_mean_div<<<(int)((tU + 255) / 256), 256, 0, stream>>>(aggU, cntU, tU);
        }

        // 4) SAGE:  sage = mean @ Wl^T + b + x_dst @ Wr^T
        launch_gemm(aggP, Wb[layer][0], xp_in, Wb[layer][2], P(layer, 1),
                    P2, N_PROD, D_FEAT, dout, 0, stream);
        launch_gemm(aggU, Wb[layer][3], xu_in, Wb[layer][5], P(layer, 4),
                    U2, N_USER, D_FEAT, dout, 0, stream);

        // 5) per-relation transform (+ReLU except last layer)
        float* outP = (layer == 2) ? outP_final : P0;
        float* outU = (layer == 2) ? outU_final : U0;
        launch_gemm(P2, Wb[layer][6], nullptr, nullptr, P(layer, 7),
                    outP, N_PROD, dout, dout, relu, stream);
        launch_gemm(U2, Wb[layer][8], nullptr, nullptr, P(layer, 9),
                    outU, N_USER, dout, dout, relu, stream);

        xu_in = U0;
        xp_in = P0;
    }
}
